// Siamese_SSTD_GCN_MLPs_only_reg_44710609551916
// MI455X (gfx1250) — compile-verified, matrix operand order checked by reading, not run on hardware
//
#include <hip/hip_runtime.h>
#include <hip/hip_bf16.h>

// ---------------------------------------------------------------------------
// Siamese ST-GCN inference on gfx1250 (MI455X).
// bf16 activations/weights, f32 accumulation via v_wmma_f32_16x16x32_bf16.
// SGC(1x1 conv) + graph conv fused into one GEMM via precomputed W2.
// Temporal 9-tap conv as 9 accumulated implicit GEMMs.
// SGC GEMM uses double-buffered GLOBAL_LOAD_ASYNC_TO_LDS_B128 staging
// (ASYNCcnt path) so global->LDS traffic overlaps the WMMA pipeline.
// ---------------------------------------------------------------------------

typedef __attribute__((ext_vector_type(16))) __bf16 v16bf;
typedef __attribute__((ext_vector_type(8)))  float  v8f;

union Frag16 {
  v16bf v;
  unsigned short u[16];
  unsigned int   w[8];
};

__device__ __forceinline__ unsigned short f2bf(float x) {
  unsigned int b = __float_as_uint(x);
  b += 0x7FFFu + ((b >> 16) & 1u);          // round-to-nearest-even
  return (unsigned short)(b >> 16);
}

__device__ __forceinline__ float silu_f(float x) {
  return x / (1.0f + __expf(-x));
}

// k offset of VGPR-pair i for half-lane h, per CDNA5 16-bit A/B matrix layout:
// i<4 -> k0 = 2*i + 8*h ; i>=4 -> k0 = 16 + 2*(i-4) + 8*h
__device__ __forceinline__ int frag_k0(int i, int h) {
  return ((i & 4) << 2) + ((i & 3) << 1) + (h << 3);
}

// Load one 16(MN) x 32(K) bf16 fragment from a row-major ushort slab.
__device__ __forceinline__ v16bf load_frag(const unsigned short* base, int stride, int lane) {
  Frag16 f;
  int l = lane & 15, h = lane >> 4;
  const unsigned short* rp = base + l * stride;
#pragma unroll
  for (int i = 0; i < 8; ++i)
    f.w[i] = *(const unsigned int*)(rp + frag_k0(i, h));
  return f.v;
}

// Per-lane async copy of 16 bytes: global memory -> LDS (GV mode).
// lds_off is the zero-based LDS byte offset (low 32 bits of the generic ptr).
__device__ __forceinline__ void async_copy16(unsigned lds_off, const void* gptr) {
  unsigned long long ga = (unsigned long long)gptr;
  asm volatile("global_load_async_to_lds_b128 %0, %1, off"
               :: "v"(lds_off), "v"(ga) : "memory");
}

__device__ __forceinline__ void wait_async0() {
  asm volatile("s_wait_asynccnt 0" ::: "memory");
}

// ------------------------- elementwise / prep kernels -----------------------

// bn0 over (v*C+ci) channels, pack x:(N,3,300,25) -> X0 bf16 [N*300][96] (cols>=75 zero)
__global__ void bn0_pack_kernel(const float* __restrict__ x,
                                const float* __restrict__ g, const float* __restrict__ b,
                                const float* __restrict__ m, const float* __restrict__ vv,
                                unsigned short* __restrict__ X0) {
  int idx = blockIdx.x * blockDim.x + threadIdx.x;
  const int TOT = 38400 * 96;
  if (idx >= TOT) return;
  int row = idx / 96, col = idx - row * 96;
  unsigned short o = 0;
  if (col < 75) {
    int ci = col / 25, v = col - ci * 25;
    int n = row / 300, t = row - n * 300;
    float xv = x[(((size_t)n * 3 + ci) * 300 + t) * 25 + v];
    int ch = v * 3 + ci;                                  // reshape is (N, V*C, T)
    float s = g[ch] * rsqrtf(vv[ch] + 1e-5f);
    o = f2bf((xv - m[ch]) * s + b[ch]);
  }
  X0[idx] = o;
}

// fold BN into y*s + t
__global__ void bn_fold_kernel(const float* __restrict__ g, const float* __restrict__ b,
                               const float* __restrict__ m, const float* __restrict__ v,
                               float* __restrict__ s, float* __restrict__ t, int C) {
  int i = blockIdx.x * blockDim.x + threadIdx.x;
  if (i < C) {
    float sc = g[i] * rsqrtf(v[i] + 1e-5f);
    s[i] = sc;
    t[i] = b[i] - m[i] * sc;
  }
}

// W2T[col=(c*25+w)][k=(ci*25+v)] = sum_k w_sgc[(k*cout+c),ci] * (A*M)[k,v,w], bf16, zero-padded
__global__ void fuse_w2t_kernel(const float* __restrict__ w_sgc,
                                const float* __restrict__ A, const float* __restrict__ Mm,
                                unsigned short* __restrict__ W2T,
                                int cin, int cout, int Kd, int NP) {
  int idx = blockIdx.x * blockDim.x + threadIdx.x;
  if (idx >= NP * Kd) return;
  int colN = idx / Kd, k = idx - colN * Kd;
  float sum = 0.f;
  if (colN < cout * 25 && k < cin * 25) {
    int c = colN / 25, w = colN - c * 25;
    int ci = k / 25,   v = k - ci * 25;
#pragma unroll
    for (int kk = 0; kk < 3; ++kk) {
      int aij = (kk * 25 + v) * 25 + w;
      sum += w_sgc[(kk * cout + c) * cin + ci] * (A[aij] * Mm[aij]);
    }
  }
  W2T[(size_t)colN * Kd + k] = f2bf(sum);
}

// bias2[c*25+w] = sum_{k,v} b_sgc[k*cout+c] * (A*M)[k,v,w]
__global__ void fuse_bias2_kernel(const float* __restrict__ b_sgc,
                                  const float* __restrict__ A, const float* __restrict__ Mm,
                                  float* __restrict__ bias2, int cout, int NP) {
  int col = blockIdx.x * blockDim.x + threadIdx.x;
  if (col >= NP) return;
  float s = 0.f;
  if (col < cout * 25) {
    int c = col / 25, w = col - c * 25;
    for (int kk = 0; kk < 3; ++kk)
      for (int v = 0; v < 25; ++v) {
        int aij = (kk * 25 + v) * 25 + w;
        s += b_sgc[kk * cout + c] * (A[aij] * Mm[aij]);
      }
  }
  bias2[col] = s;
}

// w_t (cout,cin,9,1) f32 -> WtT[dt][co][ci] bf16
__global__ void wt_pack_kernel(const float* __restrict__ w_t,
                               unsigned short* __restrict__ WtT, int C) {
  int idx = blockIdx.x * blockDim.x + threadIdx.x;
  if (idx >= 9 * C * C) return;
  int dt = idx / (C * C), r = idx - dt * C * C;
  int co = r / C, ci = r - co * C;
  WtT[idx] = f2bf(w_t[((size_t)co * C + ci) * 9 + dt]);
}

// ------------------------------ SGC mega-GEMM -------------------------------
// Y[(n,t)][(c,w)] = X[(n,t)][(ci,v)] @ W2T^T ; epilogue: +bias2, BN1, SiLU,
// scatter-store bf16 into H[(n*T+t)*25+w][cout].
// Double-buffered async global->LDS staging: chunk k+1 streams into the
// alternate LDS buffer while WMMAs consume chunk k.  80B LDS row pitch keeps
// the B128 async targets 16B-aligned and the ds fragment reads conflict-free.
__global__ void sgc_gemm_kernel(const unsigned short* __restrict__ X,
                                const unsigned short* __restrict__ W2T,
                                const float* __restrict__ bias2,
                                const float* __restrict__ s1, const float* __restrict__ t1,
                                unsigned short* __restrict__ H,
                                int M, int Kd, int coutV, int cout) {
  __shared__ __align__(16) unsigned short lA[2][64][40];
  __shared__ __align__(16) unsigned short lB[2][64][40];
  const int tid = threadIdx.x;               // 128 threads = 4 waves
  const int wave = tid >> 5, lane = tid & 31;
  const int mBase = blockIdx.x * 64;
  const int nBase = blockIdx.y * 64;

  v8f acc[4] = {};

  // async-stage one 64x32 bf16 tile of X and of W2T into LDS buffer `buf`
  auto stage = [&](int buf, int k0) {
    for (int i = tid; i < 256; i += 128) {       // 256 x 16B segments / matrix
      int r = i >> 2, s4 = (i & 3) * 8;          // row, ushort offset in row
      async_copy16((unsigned)(unsigned long long)(const void*)&lA[buf][r][s4],
                   X + (size_t)(mBase + r) * Kd + k0 + s4);
      async_copy16((unsigned)(unsigned long long)(const void*)&lB[buf][r][s4],
                   W2T + (size_t)(nBase + r) * Kd + k0 + s4);
    }
  };

  const int nChunks = Kd >> 5;
  stage(0, 0);
  int cur = 0;
  for (int kc = 0; kc < nChunks; ++kc) {
    wait_async0();          // my async writes into buf[cur] have landed in LDS
    __syncthreads();        // ...and so have everyone else's
    if (kc + 1 < nChunks) stage(cur ^ 1, (kc + 1) << 5);

    v16bf a = load_frag(&lA[cur][wave * 16][0], 40, lane);
#pragma unroll
    for (int nt = 0; nt < 4; ++nt) {
      v16bf bb = load_frag(&lB[cur][nt * 16][0], 40, lane);
      acc[nt] = __builtin_amdgcn_wmma_f32_16x16x32_bf16(
          false, a, false, bb, (short)0, acc[nt], false, false);
    }
    cur ^= 1;
  }

  // epilogue: lane(l,h), VGPR r -> tile row = h*8+r, tile col = l
  const int l = lane & 15, h = lane >> 4;
#pragma unroll
  for (int nt = 0; nt < 4; ++nt) {
    int col = nBase + nt * 16 + l;
    if (col >= coutV) continue;
    int c = col / 25, w = col - c * 25;
    float sc = s1[c], sh = t1[c], bz = bias2[col];
#pragma unroll
    for (int r = 0; r < 8; ++r) {
      int mrow = mBase + wave * 16 + h * 8 + r;
      float y = silu_f((acc[nt][r] + bz) * sc + sh);
      H[((size_t)mrow * 25 + w) * cout + c] = f2bf(y);
    }
  }
}

// --------------------------- temporal conv GEMM -----------------------------
// out[(n,to,v)][co] = sum_{dt,ci} H[(n, to*s+dt-4, v)][ci] * WtT[dt][co][ci]
__global__ void tconv_kernel(const unsigned short* __restrict__ H,
                             const unsigned short* __restrict__ WtT,
                             const float* __restrict__ bt,
                             const float* __restrict__ s2, const float* __restrict__ t2,
                             void* __restrict__ Out,
                             int Tin, int Tout, int stride, int C, int KdNext, int finalF32) {
  const int V = 25;
  const int tid = threadIdx.x, wave = tid >> 5, lane = tid & 31;
  const int l = lane & 15, h = lane >> 4;
  const long rowBase = (long)blockIdx.x * 64 + wave * 16;
  const int nTiles = C >> 4;                 // 2 or 4

  // this lane supplies A-matrix row m = l
  long ra = rowBase + l;
  int n  = (int)(ra / (Tout * V));
  int rm = (int)(ra - (long)n * Tout * V);
  int to = rm / V, v = rm - to * V;

  v8f acc[4] = {};

  for (int dt = 0; dt < 9; ++dt) {
    int ti = to * stride + dt - 4;
    bool ok = (ti >= 0) && (ti < Tin);
    const unsigned short* rp =
        ok ? (H + (((size_t)n * Tin + ti) * V + v) * C) : nullptr;
    for (int kc = 0; kc < C; kc += 32) {
      Frag16 fa;
      if (ok) {
#pragma unroll
        for (int i = 0; i < 8; ++i)
          fa.w[i] = *(const unsigned int*)(rp + kc + frag_k0(i, h));
      } else {
#pragma unroll
        for (int i = 0; i < 8; ++i) fa.w[i] = 0u;
      }
#pragma unroll
      for (int nt = 0; nt < 4; ++nt) {
        if (nt >= nTiles) break;
        Frag16 fb;
        const unsigned short* bp = WtT + ((size_t)(dt * C) + nt * 16 + l) * C + kc;
#pragma unroll
        for (int i = 0; i < 8; ++i)
          fb.w[i] = *(const unsigned int*)(bp + frag_k0(i, h));
        acc[nt] = __builtin_amdgcn_wmma_f32_16x16x32_bf16(
            false, fa.v, false, fb.v, (short)0, acc[nt], false, false);
      }
    }
  }

  // epilogue: +b_t, BN2, SiLU; store bf16 for the next block's GEMM operand,
  // or f32 for the final global-average pool.
#pragma unroll
  for (int rr = 0; rr < 8; ++rr) {
    long mr = rowBase + h * 8 + rr;
    int n2  = (int)(mr / (Tout * V));
    int rm2 = (int)(mr - (long)n2 * Tout * V);
    int to2 = rm2 / V, v2 = rm2 - to2 * V;
    for (int nt = 0; nt < nTiles; ++nt) {
      int co = nt * 16 + l;
      float y = silu_f((acc[nt][rr] + bt[co]) * s2[co] + t2[co]);
      if (finalF32)
        ((float*)Out)[(size_t)mr * C + co] = y;
      else
        ((unsigned short*)Out)[((size_t)n2 * Tout + to2) * KdNext + co * V + v2] = f2bf(y);
    }
  }
}

// ------------------------------ pool & score --------------------------------

__global__ void pool_kernel(const float* __restrict__ Hf, float* __restrict__ y, int TV) {
  int idx = blockIdx.x * blockDim.x + threadIdx.x;
  if (idx >= 128 * 64) return;
  int n = idx / 64, co = idx - n * 64;
  float s = 0.f;
  for (int f = 0; f < TV; ++f) s += Hf[((size_t)n * TV + f) * 64 + co];
  y[idx] = s / (float)TV;
}

__global__ void score_kernel(const float* __restrict__ y1, const float* __restrict__ y2,
                             const float* __restrict__ alpha, const float* __restrict__ beta,
                             float* __restrict__ out) {
  int n = blockIdx.x * blockDim.x + threadIdx.x;
  if (n >= 128) return;
  float dot = 0.f, q1 = 0.f, q2 = 0.f, dd = 0.f;
  for (int c = 0; c < 64; ++c) {
    float a = y1[n * 64 + c], b = y2[n * 64 + c];
    dot += a * b; q1 += a * a; q2 += b * b;
    float d = a - b + 1e-6f;
    dd += d * d;
  }
  float n1 = fmaxf(sqrtf(q1), 1e-6f), n2 = fmaxf(sqrtf(q2), 1e-6f);
  float cosv = dot / (n1 * n2);
  float dist = sqrtf(dd);
  float sc = alpha[0] * cosv + beta[0] * (1.f / (1.f + dist));
  out[n] = 1.f / (1.f + __expf(-sc));
}

// ------------------------------- host driver --------------------------------

extern "C" void kernel_launch(void* const* d_in, const int* in_sizes, int n_in,
                              void* d_out, int out_size, void* d_ws, size_t ws_size,
                              hipStream_t stream) {
  (void)in_sizes; (void)n_in; (void)out_size; (void)ws_size;

  // input flattening order: x1, std_x, A, bn0(g,b,m,v), 6 x block(13), alpha, beta
  const float* Aadj = (const float*)d_in[2];
  const float* bn0g = (const float*)d_in[3];
  const float* bn0b = (const float*)d_in[4];
  const float* bn0m = (const float*)d_in[5];
  const float* bn0v = (const float*)d_in[6];
  const float* alpha = (const float*)d_in[85];
  const float* beta  = (const float*)d_in[86];

  // workspace bump allocator
  char* ws = (char*)d_ws;
  size_t off = 0;
  auto alloc = [&](size_t bytes) -> char* {
    char* p = ws + off;
    off = (off + bytes + 255) & ~(size_t)255;
    return p;
  };
  unsigned short* X0  = (unsigned short*)alloc(38400ull * 1600 * 2);
  unsigned short* X1  = (unsigned short*)alloc(38400ull * 1600 * 2);
  unsigned short* Hm  = (unsigned short*)alloc(960000ull * 64 * 2);
  unsigned short* W2T = (unsigned short*)alloc(1664ull * 1600 * 2);
  float* bias2 = (float*)alloc(1664 * 4);
  unsigned short* WtT = (unsigned short*)alloc(9ull * 64 * 64 * 2);
  float* s1 = (float*)alloc(64 * 4);
  float* t1 = (float*)alloc(64 * 4);
  float* s2 = (float*)alloc(64 * 4);
  float* t2 = (float*)alloc(64 * 4);
  float* Y  = (float*)alloc(2ull * 128 * 64 * 4);
  float* Hf = (float*)X0;  // X0 is free by the last block's temporal conv

  struct Blk { int cin, cout, stride, Tin, Tout, Kd, NP; };
  const Blk BL[6] = {
    { 3, 32, 1, 300, 300,   96,  832},
    {32, 32, 1, 300, 300,  800,  832},
    {32, 32, 1, 300, 300,  800,  832},
    {32, 64, 2, 300, 150,  800, 1600},
    {64, 64, 1, 150, 150, 1600, 1600},
    {64, 64, 1, 150, 150, 1600, 1600},
  };
  unsigned short* bufs[2] = {X0, X1};

  for (int pass = 0; pass < 2; ++pass) {
    const float* xin = (const float*)d_in[pass];

    bn0_pack_kernel<<<(38400 * 96 + 255) / 256, 256, 0, stream>>>(
        xin, bn0g, bn0b, bn0m, bn0v, X0);

    for (int b = 0; b < 6; ++b) {
      const Blk& B = BL[b];
      int base = 7 + 13 * b;
      const float* w_sgc = (const float*)d_in[base + 0];
      const float* b_sgc = (const float*)d_in[base + 1];
      const float* Mm    = (const float*)d_in[base + 2];
      const float* bn1g  = (const float*)d_in[base + 3];
      const float* bn1b  = (const float*)d_in[base + 4];
      const float* bn1m  = (const float*)d_in[base + 5];
      const float* bn1v  = (const float*)d_in[base + 6];
      const float* w_t   = (const float*)d_in[base + 7];
      const float* b_t   = (const float*)d_in[base + 8];
      const float* bn2g  = (const float*)d_in[base + 9];
      const float* bn2b  = (const float*)d_in[base + 10];
      const float* bn2m  = (const float*)d_in[base + 11];
      const float* bn2v  = (const float*)d_in[base + 12];

      bn_fold_kernel<<<1, 64, 0, stream>>>(bn1g, bn1b, bn1m, bn1v, s1, t1, B.cout);
      bn_fold_kernel<<<1, 64, 0, stream>>>(bn2g, bn2b, bn2m, bn2v, s2, t2, B.cout);
      fuse_w2t_kernel<<<(B.NP * B.Kd + 255) / 256, 256, 0, stream>>>(
          w_sgc, Aadj, Mm, W2T, B.cin, B.cout, B.Kd, B.NP);
      fuse_bias2_kernel<<<(B.NP + 63) / 64, 64, 0, stream>>>(
          b_sgc, Aadj, Mm, bias2, B.cout, B.NP);
      wt_pack_kernel<<<(9 * B.cout * B.cout + 255) / 256, 256, 0, stream>>>(
          w_t, WtT, B.cout);

      int M = 128 * B.Tin;
      dim3 g1(M / 64, B.NP / 64);
      sgc_gemm_kernel<<<g1, 128, 0, stream>>>(
          bufs[b & 1], W2T, bias2, s1, t1, Hm, M, B.Kd, B.cout * 25, B.cout);

      long Mt = 128L * B.Tout * 25;
      int finalF32 = (b == 5) ? 1 : 0;
      void* outp = finalF32 ? (void*)Hf : (void*)bufs[(b + 1) & 1];
      tconv_kernel<<<(unsigned)(Mt / 64), 128, 0, stream>>>(
          Hm, WtT, b_t, s2, t2, outp, B.Tin, B.Tout, B.stride,
          B.cout, B.cout * 25, finalF32);
    }

    pool_kernel<<<(128 * 64 + 255) / 256, 256, 0, stream>>>(
        Hf, Y + (size_t)pass * 128 * 64, 150 * 25);
  }

  score_kernel<<<1, 128, 0, stream>>>(Y, Y + 128 * 64, alpha, beta, (float*)d_out);
}
